// MotifPool_3839700762837
// MI455X (gfx1250) — compile-verified, matrix operand order checked by reading, not run on hardware
//
#include <hip/hip_runtime.h>
#include <hip/hip_bf16.h>

// ---------------- problem constants ----------------
#define N_CLQv  32768
#define NEDGEv  262144
#define DIMv    256
#define HEADSv  8

typedef __bf16 bf16;
typedef __attribute__((ext_vector_type(16))) bf16     v16bf;
typedef __attribute__((ext_vector_type(8)))  float    v8f;
typedef __attribute__((ext_vector_type(4)))  unsigned u32x4;
typedef __attribute__((ext_vector_type(4)))  float    f32x4;
typedef __attribute__((ext_vector_type(4)))  int      i32x4;

union AF { v16bf v; u32x4 q[2]; };

// ---------------- async global->LDS copy (CDNA5) with safe fallback ----------------
#if defined(__gfx1250__) && __has_builtin(__builtin_amdgcn_global_load_async_to_lds_b128)
#define HAVE_ASYNC_LDS 1
typedef __attribute__((address_space(1))) i32x4 as1_i32x4;
typedef __attribute__((address_space(3))) i32x4 as3_i32x4;
#endif

__device__ __forceinline__ void async_copy16(const void* g, void* lds) {
#if defined(HAVE_ASYNC_LDS)
  __builtin_amdgcn_global_load_async_to_lds_b128((as1_i32x4*)g, (as3_i32x4*)lds, 0, 0);
#else
  *(u32x4*)lds = *(const u32x4*)g;
#endif
}
__device__ __forceinline__ void async_wait0() {
#if defined(HAVE_ASYNC_LDS)
#if __has_builtin(__builtin_amdgcn_s_wait_asynccnt)
  __builtin_amdgcn_s_wait_asynccnt(0);
#else
  asm volatile("s_wait_asynccnt 0" ::: "memory");
#endif
#endif
}

__device__ __forceinline__ bf16 f2bf(float f) {
  union { float f; unsigned u; } x; x.f = f;
  unsigned r = x.u + 0x7fffu + ((x.u >> 16) & 1u);
  union { bf16 b; unsigned short s; } o; o.s = (unsigned short)(r >> 16);
  return o.b;
}
__device__ __forceinline__ float wsum(float v) {
#pragma unroll
  for (int o = 16; o > 0; o >>= 1) v += __shfl_xor(v, o);
  return v;
}
__device__ __forceinline__ float wmaxr(float v) {
#pragma unroll
  for (int o = 16; o > 0; o >>= 1) v = fmaxf(v, __shfl_xor(v, o));
  return v;
}

// ---------------- utility kernels ----------------
__global__ void fill_zero_kernel(float* p, int n4) {
  int i = blockIdx.x * blockDim.x + threadIdx.x;
  if (i < n4) {
    f32x4 z; z[0] = 0.f; z[1] = 0.f; z[2] = 0.f; z[3] = 0.f;
    ((f32x4*)p)[i] = z;
  }
}

__global__ void f32_to_bf16_kernel(const float* __restrict__ s, bf16* __restrict__ d, int n4) {
  int i = blockIdx.x * blockDim.x + threadIdx.x;
  if (i < n4) {
    f32x4 v = ((const f32x4*)s)[i];
    int b = i * 4;
    d[b + 0] = f2bf(v[0]); d[b + 1] = f2bf(v[1]);
    d[b + 2] = f2bf(v[2]); d[b + 3] = f2bf(v[3]);
  }
}

// agg/deg -> bf16 A matrix for atom GEMM
__global__ void prep_a_kernel(const float* __restrict__ agg, const float* __restrict__ deg,
                              bf16* __restrict__ out) {
  int i = blockIdx.x * blockDim.x + threadIdx.x;  // over N_CLQ*D/4
  int idx = i * 4;
  int n = idx >> 8;
  float inv = 1.0f / fmaxf(deg[n], 1.0f);
  f32x4 v = *(const f32x4*)&agg[idx];
  out[idx + 0] = f2bf(v[0] * inv); out[idx + 1] = f2bf(v[1] * inv);
  out[idx + 2] = f2bf(v[2] * inv); out[idx + 3] = f2bf(v[3] * inv);
}

// ---------------- edge scatter: wave per edge ----------------
__global__ __launch_bounds__(256) void scatter_kernel(
    const float* __restrict__ x, const int* __restrict__ row, const int* __restrict__ col,
    float* __restrict__ agg, float* __restrict__ deg) {
  int w = threadIdx.x >> 5, l = threadIdx.x & 31;
  int e = blockIdx.x * 8 + w;
  int r = row[e], c = col[e];
  const float* xp = x + (size_t)r * DIMv + l * 8;
  float* ap = agg + (size_t)c * DIMv + l * 8;
  f32x4 a0 = *(const f32x4*)xp;
  f32x4 a1 = *(const f32x4*)(xp + 4);
#pragma unroll
  for (int u = 0; u < 4; ++u) unsafeAtomicAdd(&ap[u], a0[u]);
#pragma unroll
  for (int u = 0; u < 4; ++u) unsafeAtomicAdd(&ap[4 + u], a1[u]);
  if (l == 0) unsafeAtomicAdd(&deg[c], 1.0f);
}

// ---------------- PE branch: cx = clique_x + LN(clique_pe @ pe_w^T)  (wave per row) --------
__global__ __launch_bounds__(256) void pe_kernel(
    const float* __restrict__ clique_pe, const float* __restrict__ pe_w,
    const float* __restrict__ g, const float* __restrict__ bb,
    const float* __restrict__ clique_x, float* __restrict__ cx) {
  int w = threadIdx.x >> 5, l = threadIdx.x & 31;
  int n = blockIdx.x * 8 + w;
  float p[20];
#pragma unroll
  for (int k = 0; k < 20; ++k) p[k] = clique_pe[n * 20 + k];
  float y[8];
#pragma unroll
  for (int u = 0; u < 8; ++u) {
    int d = l * 8 + u;
    float a = 0.f;
#pragma unroll
    for (int k = 0; k < 20; ++k) a += p[k] * pe_w[d * 20 + k];
    y[u] = a;
  }
  float s = 0.f;
#pragma unroll
  for (int u = 0; u < 8; ++u) s += y[u];
  float mu = wsum(s) * (1.0f / 256.0f);
  float q = 0.f;
#pragma unroll
  for (int u = 0; u < 8; ++u) { float d = y[u] - mu; q += d * d; }
  float rinv = rsqrtf(wsum(q) * (1.0f / 256.0f) + 1e-5f);
  size_t base = (size_t)n * DIMv + l * 8;
#pragma unroll
  for (int u = 0; u < 8; ++u) {
    int d = l * 8 + u;
    cx[base + u] = clique_x[base + u] + (y[u] - mu) * rinv * g[d] + bb[d];
  }
}

// ---------------- generic WMMA GEMM: out[m][n] = sum_k A[m][k]*W[n][k] (+bias) ------------
// A: [M,256] bf16, W: [N,256] bf16.  Block: 8 waves, tile 64 rows x 256 cols.
__global__ __launch_bounds__(256) void gemm_bf16_wmma(
    const bf16* __restrict__ A, const bf16* __restrict__ W, const float* __restrict__ bias,
    float* __restrict__ outF, bf16* __restrict__ outB, int N) {
  __shared__ bf16 Alds[64 * 264];
  const int t = threadIdx.x;
  const int w = t >> 5, l = t & 31;
  const int row0 = blockIdx.x * 64;
  const int ncol0 = blockIdx.y * 256 + ((w >> 2) * 128);
  const int mrow = (w & 3) * 16;

  // stage A (64x256 bf16) to LDS (async, ASYNCcnt-tracked), padded row stride 264
#pragma unroll
  for (int i = 0; i < 8; ++i) {
    int c = t + i * 256;
    int r = c >> 5;
    int kc = (c & 31) << 3;
    async_copy16(&A[(size_t)(row0 + r) * 256 + kc], &Alds[r * 264 + kc]);
  }
  async_wait0();
  __syncthreads();

  const int nl = l & 15;
  const int kb = (l >> 4) << 3;
  v8f acc[8];
#pragma unroll
  for (int j = 0; j < 8; ++j)
#pragma unroll
    for (int i = 0; i < 8; ++i) acc[j][i] = 0.f;

  for (int kt = 0; kt < 8; ++kt) {
    AF a;
    const bf16* ap = &Alds[(mrow + nl) * 264 + kt * 32 + kb];
    a.q[0] = *(const u32x4*)(ap);
    a.q[1] = *(const u32x4*)(ap + 16);
    if (kt < 7)
      __builtin_prefetch(&W[(size_t)(ncol0 + nl) * 256 + (kt + 1) * 32], 0, 3);
#pragma unroll
    for (int j = 0; j < 8; ++j) {
      AF b;
      const bf16* wp = &W[(size_t)(ncol0 + j * 16 + nl) * 256 + kt * 32 + kb];
      b.q[0] = *(const u32x4*)(wp);
      b.q[1] = *(const u32x4*)(wp + 16);
      acc[j] = __builtin_amdgcn_wmma_f32_16x16x32_bf16(false, a.v, false, b.v,
                                                       (short)0, acc[j], false, false);
    }
  }
  const int mhi = (l >> 4) * 8;
#pragma unroll
  for (int j = 0; j < 8; ++j) {
    int n = ncol0 + j * 16 + nl;
    float bv = bias ? bias[n] : 0.f;
#pragma unroll
    for (int i = 0; i < 8; ++i) {
      int m = row0 + mrow + mhi + i;
      float v = acc[j][i] + bv;
      size_t o = (size_t)m * N + n;
      if (outF) outF[o] = v;
      if (outB) outB[o] = f2bf(v);
    }
  }
}

// ---------------- LN (+residual) wave-per-row ----------------
// mode 0: out = res + LN(src)    mode 1: out = LN(src + res)
__global__ __launch_bounds__(256) void ln_res_kernel(
    const float* __restrict__ src, const float* __restrict__ res,
    const float* __restrict__ g, const float* __restrict__ bb,
    float* __restrict__ outF, bf16* __restrict__ outB, int mode) {
  int w = threadIdx.x >> 5, l = threadIdx.x & 31;
  int n = blockIdx.x * 8 + w;
  size_t base = (size_t)n * DIMv + l * 8;
  float x[8], r[8];
#pragma unroll
  for (int u = 0; u < 8; ++u) { x[u] = src[base + u]; r[u] = res[base + u]; }
  if (mode == 1)
#pragma unroll
    for (int u = 0; u < 8; ++u) x[u] += r[u];
  float s = 0.f;
#pragma unroll
  for (int u = 0; u < 8; ++u) s += x[u];
  float mu = wsum(s) * (1.0f / 256.0f);
  float q = 0.f;
#pragma unroll
  for (int u = 0; u < 8; ++u) { float d = x[u] - mu; q += d * d; }
  float rinv = rsqrtf(wsum(q) * (1.0f / 256.0f) + 1e-5f);
#pragma unroll
  for (int u = 0; u < 8; ++u) {
    int d = l * 8 + u;
    float y = (x[u] - mu) * rinv * g[d] + bb[d];
    if (mode == 0) y += r[u];
    outF[base + u] = y;
    if (outB) outB[base + u] = f2bf(y);
  }
}

// ---------------- attention: one block per (batch, head), 4 waves ----------------
__global__ __launch_bounds__(128) void attn_kernel(const bf16* __restrict__ qkv,
                                                   bf16* __restrict__ oBf) {
  extern __shared__ char smem[];
  bf16* Klds = (bf16*)smem;                       // [256][40]
  bf16* Vt = (bf16*)(smem + 20480);               // [32][264] (V transposed)
  float* S = (float*)(smem + 20480 + 16896);      // 4 waves x [16][256] f32
  const int t = threadIdx.x, w = t >> 5, l = t & 31;
  const int b = blockIdx.x >> 3, h = blockIdx.x & 7;
  const int n0 = b * 256;

  // stage K (async to LDS) and V^T (through VGPRs; needs transpose)
#pragma unroll
  for (int rr = 0; rr < 2; ++rr) {
    int r = t + rr * 128;
    const bf16* kp = qkv + (size_t)(n0 + r) * 768 + 256 + h * 32;
    const bf16* vp = kp + 256;
#pragma unroll
    for (int ch = 0; ch < 4; ++ch) {
      async_copy16(kp + ch * 8, &Klds[r * 40 + ch * 8]);
      union { u32x4 q; bf16 e[8]; } vv;
      vv.q = *(const u32x4*)(vp + ch * 8);
#pragma unroll
      for (int u = 0; u < 8; ++u) Vt[(ch * 8 + u) * 264 + r] = vv.e[u];
    }
  }
  async_wait0();
  __syncthreads();

  float* Sw = S + w * 16 * 256;
  bf16* Pw = (bf16*)Sw;  // reuse; bf16 row stride 264 (never clobbers unread f32 rows)
  const int nl = l & 15, kb = (l >> 4) << 3, mhi = (l >> 4) * 8;
  const float scale = 0.17677669529663687f;  // 1/sqrt(32)

  for (int it = 0; it < 4; ++it) {
    int m0 = (w * 4 + it) * 16;
    AF qf;
    const bf16* qp = qkv + (size_t)(n0 + m0 + nl) * 768 + h * 32 + kb;
    qf.q[0] = *(const u32x4*)qp;
    qf.q[1] = *(const u32x4*)(qp + 16);
    // scores: 16 x 256 via 16 WMMA tiles (K=32 single step)
#pragma unroll
    for (int j = 0; j < 16; ++j) {
      AF kf;
      const bf16* kp = &Klds[(j * 16 + nl) * 40 + kb];
      kf.q[0] = *(const u32x4*)kp;
      kf.q[1] = *(const u32x4*)(kp + 16);
      v8f sc;
#pragma unroll
      for (int i = 0; i < 8; ++i) sc[i] = 0.f;
      sc = __builtin_amdgcn_wmma_f32_16x16x32_bf16(false, qf.v, false, kf.v,
                                                   (short)0, sc, false, false);
#pragma unroll
      for (int i = 0; i < 8; ++i) Sw[(mhi + i) * 256 + j * 16 + nl] = sc[i] * scale;
    }
    // wave-local softmax over each of the 16 rows, write bf16 P
    for (int r = 0; r < 16; ++r) {
      float vloc[8]; float m = -1e30f;
#pragma unroll
      for (int u = 0; u < 8; ++u) { vloc[u] = Sw[r * 256 + l * 8 + u]; m = fmaxf(m, vloc[u]); }
      m = wmaxr(m);
      float s = 0.f;
#pragma unroll
      for (int u = 0; u < 8; ++u) { vloc[u] = __expf(vloc[u] - m); s += vloc[u]; }
      float inv = 1.0f / wsum(s);
#pragma unroll
      for (int u = 0; u < 8; ++u) Pw[r * 264 + l * 8 + u] = f2bf(vloc[u] * inv);
    }
    // O tile (16x32) = P (16x256) @ V (256x32): 8 k-steps x 2 n-tiles
    v8f oa[2];
#pragma unroll
    for (int j = 0; j < 2; ++j)
#pragma unroll
      for (int i = 0; i < 8; ++i) oa[j][i] = 0.f;
    for (int kt = 0; kt < 8; ++kt) {
      AF pf;
      const bf16* pp = &Pw[nl * 264 + kt * 32 + kb];
      pf.q[0] = *(const u32x4*)pp;
      pf.q[1] = *(const u32x4*)(pp + 16);
#pragma unroll
      for (int j = 0; j < 2; ++j) {
        AF vf;
        const bf16* vtp = &Vt[(j * 16 + nl) * 264 + kt * 32 + kb];
        vf.q[0] = *(const u32x4*)vtp;
        vf.q[1] = *(const u32x4*)(vtp + 16);
        oa[j] = __builtin_amdgcn_wmma_f32_16x16x32_bf16(false, pf.v, false, vf.v,
                                                        (short)0, oa[j], false, false);
      }
    }
#pragma unroll
    for (int j = 0; j < 2; ++j)
#pragma unroll
      for (int i = 0; i < 8; ++i) {
        int n = n0 + m0 + mhi + i;
        oBf[(size_t)n * 256 + h * 32 + j * 16 + nl] = f2bf(oa[j][i]);
      }
  }
}

// ---------------- per-graph MLP score + segment softmax + pooled output ----------------
__global__ __launch_bounds__(256) void mlp_pool_kernel(
    const float* __restrict__ cx, const float* __restrict__ w1, const float* __restrict__ b1,
    const float* __restrict__ w2, const float* __restrict__ b2,
    float* __restrict__ drug, float* __restrict__ alphaOut) {
  extern __shared__ char smem[];
  float* W1 = (float*)smem;          // 16384
  float* B1 = W1 + 16384;            // 512
  float* W2 = B1 + 512;              // 512
  float* B2v = W2 + 512;             // 8
  float* srow = B2v + 8;             // 8*256
  float* scores = srow + 2048;       // 256*8
  float* mx = scores + 2048;         // 8
  float* dn = mx + 8;                // 8
  int t = threadIdx.x, w = t >> 5, l = t & 31, g = blockIdx.x;
  for (int i = t; i < 16384; i += 256) W1[i] = w1[i];
  for (int i = t; i < 512; i += 256) { B1[i] = b1[i]; W2[i] = w2[i]; }
  if (t < 8) B2v[t] = b2[t];
  __syncthreads();

  int h = l >> 2, qq = l & 3;
  for (int it = 0; it < 32; ++it) {
    int r = it * 8 + w;
    size_t base = (size_t)(g * 256 + r) * 256;
#pragma unroll
    for (int u = 0; u < 8; ++u) srow[w * 256 + l * 8 + u] = cx[base + l * 8 + u];
    float p = 0.f;
    for (int dd = 0; dd < 16; ++dd) {
      int d = qq * 16 + dd;
      float a = B1[h * 64 + d];
      const float* wr = &W1[(h * 64 + d) * 32];
      const float* sr = &srow[w * 256 + h * 32];
#pragma unroll
      for (int c = 0; c < 32; ++c) a += sr[c] * wr[c];
      p += fmaxf(a, 0.f) * W2[h * 64 + d];
    }
    p += __shfl_xor(p, 1);
    p += __shfl_xor(p, 2);
    if (qq == 0) scores[r * 8 + h] = p + B2v[h];
  }
  __syncthreads();
  {  // wave w reduces head w
    float m = -1e30f;
#pragma unroll
    for (int k2 = 0; k2 < 8; ++k2) m = fmaxf(m, scores[(l + k2 * 32) * 8 + w]);
    m = wmaxr(m);
    float s = 0.f;
#pragma unroll
    for (int k2 = 0; k2 < 8; ++k2) s += __expf(scores[(l + k2 * 32) * 8 + w] - m);
    s = wsum(s);
    if (l == 0) { mx[w] = m; dn[w] = s; }
  }
  __syncthreads();
#pragma unroll
  for (int i = 0; i < 8; ++i) {
    int idx = t + i * 256;
    int r = idx >> 3, hh = idx & 7;
    float a = __expf(scores[idx] - mx[hh]) / dn[hh];
    alphaOut[(size_t)(g * 256 + r) * 8 + hh] = a;
    scores[idx] = a;
  }
  __syncthreads();
  {  // pooled feature: thread t owns channel t
    float acc = 0.f;
    int hh = t >> 5;
    for (int r = 0; r < 256; ++r)
      acc += cx[(size_t)(g * 256 + r) * 256 + t] * scores[r * 8 + hh];
    drug[(size_t)g * 256 + t] = acc;
  }
}

// ---------------- launch ----------------
extern "C" void kernel_launch(void* const* d_in, const int* in_sizes, int n_in,
                              void* d_out, int out_size, void* d_ws, size_t ws_size,
                              hipStream_t stream) {
  const float* x         = (const float*)d_in[0];
  const float* clique_x  = (const float*)d_in[1];
  const float* clique_pe = (const float*)d_in[2];
  const int*   rowi      = (const int*)d_in[3];
  const int*   coli      = (const int*)d_in[4];
  // d_in[5] = clique_batch (contiguous equal counts -> implicit)
  const float* pe_w   = (const float*)d_in[6];
  const float* pe_g   = (const float*)d_in[7];
  const float* pe_b   = (const float*)d_in[8];
  const float* atom_w = (const float*)d_in[9];
  const float* atom_g = (const float*)d_in[10];
  const float* atom_b = (const float*)d_in[11];
  const float* in_w   = (const float*)d_in[12];
  const float* in_b   = (const float*)d_in[13];
  const float* out_w  = (const float*)d_in[14];
  const float* out_b  = (const float*)d_in[15];
  const float* cn_g   = (const float*)d_in[16];
  const float* cn_b   = (const float*)d_in[17];
  const float* mlp_w1 = (const float*)d_in[18];
  const float* mlp_b1 = (const float*)d_in[19];
  const float* mlp_w2 = (const float*)d_in[20];
  const float* mlp_b2 = (const float*)d_in[21];

  float* drug  = (float*)d_out;                 // [128*256]
  float* cxp   = drug + 32768;                  // [32768*256]
  float* alphp = cxp + 8388608;                 // [32768*8]

  char* ws = (char*)d_ws;
  float* agg = (float*)(ws + 0);                // 33554432 B
  float* deg = (float*)(ws + 33554432);         // 131072 B
  bf16* abf  = (bf16*)(ws + 33685504);          // 16777216 B
  bf16* cxbf = (bf16*)(ws + 50462720);          // 16777216 B
  bf16* qkvb = (bf16*)(ws + 67239936);          // 50331648 B
  bf16* obf  = (bf16*)(ws + 117571584);         // 16777216 B
  float* tmp = (float*)(ws + 134348800);        // 33554432 B
  bf16* wat  = (bf16*)(ws + 167903232);         // 131072 B
  bf16* win  = (bf16*)(ws + 168034304);         // 393216 B
  bf16* wout = (bf16*)(ws + 168427520);         // 131072 B

  (void)in_sizes; (void)n_in; (void)out_size; (void)ws_size;
  (void)hipFuncSetAttribute((const void*)attn_kernel,
                            hipFuncAttributeMaxDynamicSharedMemorySize, 102912);
  (void)hipFuncSetAttribute((const void*)mlp_pool_kernel,
                            hipFuncAttributeMaxDynamicSharedMemorySize, 86112);

  // 1) zero agg+deg (contiguous)
  fill_zero_kernel<<<8224, 256, 0, stream>>>(agg, (N_CLQv * DIMv + N_CLQv) / 4);
  // 2) edge gather/scatter (wave per edge)
  scatter_kernel<<<NEDGEv / 8, 256, 0, stream>>>(x, rowi, coli, agg, deg);
  // 3) PE branch into cx (clique_x + LN(pe))
  pe_kernel<<<N_CLQv / 8, 256, 0, stream>>>(clique_pe, pe_w, pe_g, pe_b, clique_x, cxp);
  // 4) A = agg/deg -> bf16; weights -> bf16
  prep_a_kernel<<<N_CLQv * DIMv / 4 / 256, 256, 0, stream>>>(agg, deg, abf);
  f32_to_bf16_kernel<<<64, 256, 0, stream>>>(atom_w, wat, 65536 / 4);
  f32_to_bf16_kernel<<<192, 256, 0, stream>>>(in_w, win, 196608 / 4);
  f32_to_bf16_kernel<<<64, 256, 0, stream>>>(out_w, wout, 65536 / 4);
  // 5) atom GEMM -> tmp; cx = cx + LN(tmp), bf16 copy for qkv GEMM
  gemm_bf16_wmma<<<dim3(N_CLQv / 64, 1), 256, 0, stream>>>(abf, wat, nullptr, tmp, nullptr, 256);
  ln_res_kernel<<<N_CLQv / 8, 256, 0, stream>>>(tmp, cxp, atom_g, atom_b, cxp, cxbf, 0);
  // 6) qkv GEMM (N=768, +bias) -> bf16 qkv
  gemm_bf16_wmma<<<dim3(N_CLQv / 64, 3), 256, 0, stream>>>(cxbf, win, in_b, nullptr, qkvb, 768);
  // 7) attention per (batch, head)
  attn_kernel<<<128 * HEADSv, 128, 102912, stream>>>(qkvb, obf);
  // 8) out projection (+bias) -> tmp; cx = LN(cx + tmp)
  gemm_bf16_wmma<<<dim3(N_CLQv / 64, 1), 256, 0, stream>>>(obf, wout, out_b, tmp, nullptr, 256);
  ln_res_kernel<<<N_CLQv / 8, 256, 0, stream>>>(tmp, cxp, cn_g, cn_b, cxp, nullptr, 1);
  // 9) per-graph MLP score + segment softmax + pooled drug feature
  mlp_pool_kernel<<<128, 256, 86112, stream>>>(cxp, mlp_w1, mlp_b1, mlp_w2, mlp_b2, drug, alphp);
}
